// PathologyCLIP_89988154785989
// MI455X (gfx1250) — matrix operand order repaired and checked by reading, not compiled, not run on hardware
//
#include <hip/hip_runtime.h>

#define B_SZ   32
#define P_SZ   2048
#define D_SZ   1024
#define T_SZ   768
#define PR_SZ  512
#define M_ROWS (B_SZ * P_SZ)   // 65536
#define PTILES (P_SZ / 16)     // 128

typedef __bf16 bf16;
typedef __attribute__((ext_vector_type(16))) __bf16 v16bf;
typedef __attribute__((ext_vector_type(8)))  float  v8f;

static __device__ __forceinline__ float bf2f(bf16 h) { return (float)h; }
static __device__ __forceinline__ bf16  f2bf(float f) { return (bf16)f; }

// ---------------------------------------------------------------------------
// fp32 -> bf16 row-major convert
// ---------------------------------------------------------------------------
__global__ void cvt_bf16_kernel(const float* __restrict__ src,
                                bf16* __restrict__ dst, long n) {
    long i = (long)blockIdx.x * blockDim.x + threadIdx.x;
    long stride = (long)gridDim.x * blockDim.x;
    for (; i < n; i += stride) dst[i] = f2bf(src[i]);
}

// ---------------------------------------------------------------------------
// Pack W [K][N] fp32 into per-lane contiguous bf16 B-fragments:
// dst[(((nt*(K/32) + kt)*32) + lane)*16 + e] = W[kt*32 + (lane/16)*16 + e][nt*16 + lane%16]
// ---------------------------------------------------------------------------
__global__ void pack_b_kernel(const float* __restrict__ W,
                              bf16* __restrict__ dst, int K, int N) {
    const int Kt = K >> 5;
    long total = (long)K * N;
    long i = (long)blockIdx.x * blockDim.x + threadIdx.x;
    long stride = (long)gridDim.x * blockDim.x;
    for (; i < total; i += stride) {
        int e  = (int)(i & 15);
        int l  = (int)((i >> 4) & 31);
        long r = i >> 9;
        int kt = (int)(r % Kt);
        int nt = (int)(r / Kt);
        int kk = kt * 32 + (l >> 4) * 16 + e;
        int n  = nt * 16 + (l & 15);
        dst[i] = f2bf(W[(long)kk * N + n]);
    }
}

// ---------------------------------------------------------------------------
// bf16 WMMA GEMM: C[M][N] = A[M][K] @ W + bias, C stored bf16.
// Block = 4 waves (128 thr). Block tile 64(M) x 64(N); each wave 16x64.
// Ping-pong (unroll-2) fragment buffers: no register copies, loads for the
// next k-step stay in flight across the current step's WMMAs.
// K must be a multiple of 64 (Kt even).
// ---------------------------------------------------------------------------
__global__ __launch_bounds__(128)
void gemm_bf16_kernel(const bf16* __restrict__ A, const bf16* __restrict__ Bpk,
                      const float* __restrict__ bias, bf16* __restrict__ C,
                      int M, int N, int K) {
    const int tid  = threadIdx.x;
    const int wave = tid >> 5;
    const int lane = tid & 31;
    const int lo   = lane & 15;
    const int hi   = lane >> 4;
    const int mbase = blockIdx.x * 64 + wave * 16;
    const int nbase = blockIdx.y * 64;
    const int Kt    = K >> 5;
    const int ntg0  = nbase >> 4;

    const bf16* arow = A + (long)(mbase + lo) * K + 16 * hi;

    auto load_a = [&](int kt) -> v16bf {
        const bf16* ap = arow + kt * 32;
        v16bf a;
        #pragma unroll
        for (int e = 0; e < 16; ++e) a[e] = ap[e];
        return a;
    };
    auto load_b = [&](int j, int kt) -> v16bf {
        const bf16* bp = Bpk + ((((long)(ntg0 + j) * Kt) + kt) * 32 + lane) * 16;
        v16bf b;
        #pragma unroll
        for (int e = 0; e < 16; ++e) b[e] = bp[e];
        return b;
    };

    v8f acc[4];
    #pragma unroll
    for (int j = 0; j < 4; ++j) acc[j] = (v8f){0,0,0,0,0,0,0,0};

    v16bf a0 = load_a(0);
    v16bf b0[4];
    #pragma unroll
    for (int j = 0; j < 4; ++j) b0[j] = load_b(j, 0);

    for (int kt = 0; kt < Kt; kt += 2) {
        if (kt + 2 < Kt) __builtin_prefetch(arow + (kt + 2) * 32, 0, 1);
        // load odd-step buffers while even-step WMMAs run
        v16bf a1 = load_a(kt + 1);
        v16bf b1[4];
        #pragma unroll
        for (int j = 0; j < 4; ++j) b1[j] = load_b(j, kt + 1);
        #pragma unroll
        for (int j = 0; j < 4; ++j)
            acc[j] = __builtin_amdgcn_wmma_f32_16x16x32_bf16(
                false, a0, false, b0[j], (short)0, acc[j], false, false);
        // load next even-step buffers while odd-step WMMAs run
        if (kt + 2 < Kt) {
            a0 = load_a(kt + 2);
            #pragma unroll
            for (int j = 0; j < 4; ++j) b0[j] = load_b(j, kt + 2);
        }
        #pragma unroll
        for (int j = 0; j < 4; ++j)
            acc[j] = __builtin_amdgcn_wmma_f32_16x16x32_bf16(
                false, a1, false, b1[j], (short)0, acc[j], false, false);
    }

    #pragma unroll
    for (int j = 0; j < 4; ++j) {
        const int col = nbase + j * 16 + lo;
        const float bc2 = bias[col];
        #pragma unroll
        for (int r = 0; r < 8; ++r) {
            const int row = mbase + 8 * hi + r;
            C[(long)row * N + col] = f2bf(acc[j][r] + bc2);
        }
    }
}

// ---------------------------------------------------------------------------
// Fused scores + softmax + column-sum kernel.
// Block = (batch b, 16-row p-tile). 4 waves; wave w owns columns [w*512, w*512+512),
// processed as 8 groups of 4 N-tiles (A fragment reused 4x, ping-pong buffered).
// Builds the 16 x 2048 score strip in dynamic LDS via WMMA, softmaxes each row,
// then writes deterministic per-tile column sums of attn into `part`.
// ---------------------------------------------------------------------------
__global__ __launch_bounds__(128)
void attn_kernel(const bf16* __restrict__ qb, const bf16* __restrict__ kb,
                 float* __restrict__ part /* [B][PTILES][P] */) {
    extern __shared__ float smem[];            // 16 * 2048 floats
    __shared__ float sred[128];
    __shared__ float rowmax[16], rowinv[16];

    const int tid  = threadIdx.x;
    const int wave = tid >> 5;
    const int lane = tid & 31;
    const int lo   = lane & 15;
    const int hi   = lane >> 4;
    const int b    = blockIdx.y;
    const int pt   = blockIdx.x;
    const int p0   = pt * 16;
    const float scale = 0.03125f;              // 1/sqrt(1024)

    const bf16* qrow = qb + ((long)(b * P_SZ + p0 + lo)) * D_SZ + 16 * hi;

    auto load_a = [&](int kt) -> v16bf {
        const bf16* ap = qrow + kt * 32;
        v16bf a;
        #pragma unroll
        for (int e = 0; e < 16; ++e) a[e] = ap[e];
        return a;
    };

    for (int grp = 0; grp < 8; ++grp) {
        const int n0 = wave * 512 + grp * 64;
        const bf16* kbase = kb + ((long)(b * P_SZ + n0 + lo)) * D_SZ + 16 * hi;

        auto load_b = [&](int j, int kt) -> v16bf {
            const bf16* bp = kbase + (long)j * 16 * D_SZ + kt * 32;
            v16bf v;
            #pragma unroll
            for (int e = 0; e < 16; ++e) v[e] = bp[e];
            return v;
        };

        v8f acc[4];
        #pragma unroll
        for (int j = 0; j < 4; ++j) acc[j] = (v8f){0,0,0,0,0,0,0,0};

        v16bf a0 = load_a(0);
        v16bf b0[4];
        #pragma unroll
        for (int j = 0; j < 4; ++j) b0[j] = load_b(j, 0);

        for (int kt = 0; kt < 32; kt += 2) {
            if (kt + 2 < 32) __builtin_prefetch(qrow + (kt + 2) * 32, 0, 1);
            v16bf a1 = load_a(kt + 1);
            v16bf b1[4];
            #pragma unroll
            for (int j = 0; j < 4; ++j) b1[j] = load_b(j, kt + 1);
            #pragma unroll
            for (int j = 0; j < 4; ++j)
                acc[j] = __builtin_amdgcn_wmma_f32_16x16x32_bf16(
                    false, a0, false, b0[j], (short)0, acc[j], false, false);
            if (kt + 2 < 32) {
                a0 = load_a(kt + 2);
                #pragma unroll
                for (int j = 0; j < 4; ++j) b0[j] = load_b(j, kt + 2);
            }
            #pragma unroll
            for (int j = 0; j < 4; ++j)
                acc[j] = __builtin_amdgcn_wmma_f32_16x16x32_bf16(
                    false, a1, false, b1[j], (short)0, acc[j], false, false);
        }

        #pragma unroll
        for (int j = 0; j < 4; ++j)
            #pragma unroll
            for (int r = 0; r < 8; ++r)
                smem[(r + 8 * hi) * P_SZ + n0 + j * 16 + lo] = acc[j][r] * scale;
    }
    __syncthreads();

    // per-row softmax stats: 8 threads per row, 256 columns each
    const int r  = tid >> 3;
    const int c0 = (tid & 7) * 256;
    {
        float m = -1e30f;
        for (int c = 0; c < 256; ++c) m = fmaxf(m, smem[r * P_SZ + c0 + c]);
        sred[tid] = m;
        __syncthreads();
        if ((tid & 7) == 0) {
            float mm = sred[tid];
            #pragma unroll
            for (int j = 1; j < 8; ++j) mm = fmaxf(mm, sred[tid + j]);
            rowmax[r] = mm;
        }
        __syncthreads();
        const float mrow = rowmax[r];
        float s = 0.f;
        for (int c = 0; c < 256; ++c) s += __expf(smem[r * P_SZ + c0 + c] - mrow);
        sred[tid] = s;
        __syncthreads();
        if ((tid & 7) == 0) {
            float ss = 0.f;
            #pragma unroll
            for (int j = 0; j < 8; ++j) ss += sred[tid + j];
            rowinv[r] = 1.0f / ss;
        }
        __syncthreads();
    }

    // deterministic per-tile column sums of attn rows
    float* pout = part + ((long)(b * PTILES + pt)) * P_SZ;
    for (int c = tid; c < P_SZ; c += 128) {
        float s = 0.f;
        #pragma unroll
        for (int rr = 0; rr < 16; ++rr)
            s += __expf(smem[rr * P_SZ + c] - rowmax[rr]) * rowinv[rr];
        pout[c] = s;
    }
}

// attn_weights[b][c] = (1/P) * sum_pt part[b][pt][c]   (fixed order => deterministic)
__global__ void attn_reduce_kernel(const float* __restrict__ part,
                                   float* __restrict__ out_attn) {
    const int b = blockIdx.y;
    const int c = blockIdx.x * blockDim.x + threadIdx.x;
    const float* p = part + (long)b * PTILES * P_SZ + c;
    float s = 0.f;
    for (int t = 0; t < PTILES; ++t) s += p[(long)t * P_SZ];
    out_attn[b * P_SZ + c] = s * (1.0f / P_SZ);
}

// weighted[b][d] = sum_q attn_weights[b][q] * v[b][q][d]
__global__ __launch_bounds__(256)
void weighted_kernel(const float* __restrict__ aw, const bf16* __restrict__ vb,
                     float* __restrict__ weighted) {
    const int b = blockIdx.x;
    for (int d = threadIdx.x; d < D_SZ; d += 256) {
        const bf16* vp = vb + (long)b * P_SZ * D_SZ + d;
        const float* ap = aw + b * P_SZ;
        float s = 0.f;
        for (int q = 0; q < P_SZ; ++q) s += ap[q] * bf2f(vp[(long)q * D_SZ]);
        weighted[b * D_SZ + d] = s;
    }
}

// ---------------------------------------------------------------------------
// MLP encoder: relu(x@W1+b1) -> layernorm(g,bt) -> relu(@W2+b2) -> L2 norm
// One block per row (256 threads). din <= 1024, hidden 1024, out 512.
// ---------------------------------------------------------------------------
__global__ __launch_bounds__(256)
void mlp_kernel(const float* __restrict__ X, int xstride, int din,
                const float* __restrict__ W1, const float* __restrict__ b1,
                const float* __restrict__ g,  const float* __restrict__ bt,
                const float* __restrict__ W2, const float* __restrict__ b2,
                float* __restrict__ out) {
    __shared__ float xs[1024];
    __shared__ float hs[1024];
    __shared__ float red[256];
    const int row = blockIdx.x;
    const int tid = threadIdx.x;
    const float* x = X + (long)row * xstride;
    for (int i = tid; i < din; i += 256) xs[i] = x[i];
    __syncthreads();

    for (int j = tid; j < 1024; j += 256) {
        float s = b1[j];
        for (int k = 0; k < din; ++k) s += xs[k] * W1[(long)k * 1024 + j];
        hs[j] = fmaxf(s, 0.f);
    }
    __syncthreads();

    float ls = 0.f;
    for (int j = tid; j < 1024; j += 256) ls += hs[j];
    red[tid] = ls; __syncthreads();
    for (int off = 128; off > 0; off >>= 1) {
        if (tid < off) red[tid] += red[tid + off];
        __syncthreads();
    }
    const float mu = red[0] * (1.0f / 1024.0f); __syncthreads();

    float vs = 0.f;
    for (int j = tid; j < 1024; j += 256) { float d = hs[j] - mu; vs += d * d; }
    red[tid] = vs; __syncthreads();
    for (int off = 128; off > 0; off >>= 1) {
        if (tid < off) red[tid] += red[tid + off];
        __syncthreads();
    }
    const float rstd = rsqrtf(red[0] * (1.0f / 1024.0f) + 1e-5f); __syncthreads();

    for (int j = tid; j < 1024; j += 256) hs[j] = (hs[j] - mu) * rstd * g[j] + bt[j];
    __syncthreads();

    for (int j = tid; j < 512; j += 256) {
        float s = b2[j];
        for (int k = 0; k < 1024; ++k) s += hs[k] * W2[(long)k * 512 + j];
        xs[j] = fmaxf(s, 0.f);
    }
    __syncthreads();

    float ns = 0.f;
    for (int j = tid; j < 512; j += 256) { float v = xs[j]; ns += v * v; }
    red[tid] = ns; __syncthreads();
    for (int off = 128; off > 0; off >>= 1) {
        if (tid < off) red[tid] += red[tid + off];
        __syncthreads();
    }
    const float inv = rsqrtf(red[0]); __syncthreads();
    for (int j = tid; j < 512; j += 256) out[(long)row * 512 + j] = xs[j] * inv;
}

// logits (32x32 each) + transposes, all four output blocks
__global__ void logits_kernel(const float* __restrict__ pf,  const float* __restrict__ thf,
                              const float* __restrict__ torg, const float* __restrict__ thist,
                              const float* __restrict__ logit_scale, float* __restrict__ out) {
    const int idx = blockIdx.x * blockDim.x + threadIdx.x;
    if (idx >= B_SZ * B_SZ) return;
    const int i = idx >> 5, j = idx & 31;
    const float ls = __expf(logit_scale[0]);
    float s_thum = 0.f, s_patch = 0.f;
    for (int k = 0; k < PR_SZ; ++k) {
        s_thum  += thf[i * PR_SZ + k] * torg[j * PR_SZ + k];
        s_patch += pf[i * PR_SZ + k]  * thist[j * PR_SZ + k];
    }
    s_thum *= ls; s_patch *= ls;
    out[           i * 32 + j] = s_thum;    // logits_per_thum
    out[1024 +     i * 32 + j] = s_patch;   // logits_per_patch
    out[2048 +     j * 32 + i] = s_thum;    // logits_per_thum.T
    out[3072 +     j * 32 + i] = s_patch;   // logits_per_patch.T
}

// ---------------------------------------------------------------------------
extern "C" void kernel_launch(void* const* d_in, const int* in_sizes, int n_in,
                              void* d_out, int out_size, void* d_ws, size_t ws_size,
                              hipStream_t stream) {
    (void)in_sizes; (void)n_in; (void)out_size; (void)ws_size;
    const float* thum  = (const float*)d_in[0];
    const float* patch = (const float*)d_in[1];
    const float* text  = (const float*)d_in[2];
    const float* Wq = (const float*)d_in[3];  const float* bq = (const float*)d_in[4];
    const float* Wk = (const float*)d_in[5];  const float* bk = (const float*)d_in[6];
    const float* Wv = (const float*)d_in[7];  const float* bv = (const float*)d_in[8];
    const float* peW1 = (const float*)d_in[9];  const float* peb1 = (const float*)d_in[10];
    const float* peg  = (const float*)d_in[11]; const float* pebt = (const float*)d_in[12];
    const float* peW2 = (const float*)d_in[13]; const float* peb2 = (const float*)d_in[14];
    const float* teW1 = (const float*)d_in[15]; const float* teb1 = (const float*)d_in[16];
    const float* teg  = (const float*)d_in[17]; const float* tebt = (const float*)d_in[18];
    const float* teW2 = (const float*)d_in[19]; const float* teb2 = (const float*)d_in[20];
    const float* thW1 = (const float*)d_in[21]; const float* thb1 = (const float*)d_in[22];
    const float* thg  = (const float*)d_in[23]; const float* thbt = (const float*)d_in[24];
    const float* thW2 = (const float*)d_in[25]; const float* thb2 = (const float*)d_in[26];
    const float* logit_scale = (const float*)d_in[27];
    float* out = (float*)d_out;
    float* out_attn = out + 4096;   // attn_weights region [32][2048]

    char* ws = (char*)d_ws;
    size_t off = 0;
    auto walloc = [&](size_t bytes) -> void* {
        void* p = ws + off;
        off = (off + bytes + 255) & ~(size_t)255;
        return p;
    };
    const size_t PD = (size_t)M_ROWS * D_SZ;           // 64Mi elems
    bf16* patchb = (bf16*)walloc(PD * sizeof(bf16));
    bf16* qb     = (bf16*)walloc(PD * sizeof(bf16));
    bf16* kb     = (bf16*)walloc(PD * sizeof(bf16));
    bf16* vb     = (bf16*)walloc(PD * sizeof(bf16));
    bf16* Wq_pk  = (bf16*)walloc((size_t)D_SZ * D_SZ * sizeof(bf16));
    bf16* Wk_pk  = (bf16*)walloc((size_t)D_SZ * D_SZ * sizeof(bf16));
    bf16* Wv_pk  = (bf16*)walloc((size_t)D_SZ * D_SZ * sizeof(bf16));
    float* part  = (float*)walloc((size_t)B_SZ * PTILES * P_SZ * sizeof(float));
    float* wtd   = (float*)walloc((size_t)B_SZ * D_SZ * sizeof(float));
    float* pf    = (float*)walloc((size_t)B_SZ * PR_SZ * sizeof(float));
    float* thf   = (float*)walloc((size_t)B_SZ * PR_SZ * sizeof(float));
    float* torg  = (float*)walloc((size_t)B_SZ * PR_SZ * sizeof(float));
    float* thist = (float*)walloc((size_t)B_SZ * PR_SZ * sizeof(float));

    // 1) stage patch in bf16, pack weights into B-fragment layout
    cvt_bf16_kernel<<<8192, 256, 0, stream>>>(patch, patchb, (long)PD);
    pack_b_kernel<<<2048, 256, 0, stream>>>(Wq, Wq_pk, D_SZ, D_SZ);
    pack_b_kernel<<<2048, 256, 0, stream>>>(Wk, Wk_pk, D_SZ, D_SZ);
    pack_b_kernel<<<2048, 256, 0, stream>>>(Wv, Wv_pk, D_SZ, D_SZ);

    // 2) q/k/v projections (WMMA bf16, fp32 accumulate)
    dim3 ggrid(M_ROWS / 64, D_SZ / 64);
    gemm_bf16_kernel<<<ggrid, 128, 0, stream>>>(patchb, Wq_pk, bq, qb, M_ROWS, D_SZ, D_SZ);
    gemm_bf16_kernel<<<ggrid, 128, 0, stream>>>(patchb, Wk_pk, bk, kb, M_ROWS, D_SZ, D_SZ);
    gemm_bf16_kernel<<<ggrid, 128, 0, stream>>>(patchb, Wv_pk, bv, vb, M_ROWS, D_SZ, D_SZ);

    // 3) fused scores + softmax + column sums (WMMA bf16 -> fp32 softmax in LDS)
    attn_kernel<<<dim3(PTILES, B_SZ), 128, 16 * P_SZ * sizeof(float), stream>>>(qb, kb, part);
    attn_reduce_kernel<<<dim3(P_SZ / 256, B_SZ), 256, 0, stream>>>(part, out_attn);

    // 4) weighted = attn_weights @ v
    weighted_kernel<<<B_SZ, 256, 0, stream>>>(out_attn, vb, wtd);

    // 5) encoders (fp32 VALU, tiny)
    mlp_kernel<<<B_SZ, 256, 0, stream>>>(wtd, D_SZ, D_SZ, peW1, peb1, peg, pebt, peW2, peb2, pf);
    mlp_kernel<<<B_SZ, 256, 0, stream>>>(thum, D_SZ, D_SZ, thW1, thb1, thg, thbt, thW2, thb2, thf);
    mlp_kernel<<<B_SZ, 256, 0, stream>>>(text + T_SZ,     3 * T_SZ, T_SZ, teW1, teb1, teg, tebt, teW2, teb2, torg);
    mlp_kernel<<<B_SZ, 256, 0, stream>>>(text + 2 * T_SZ, 3 * T_SZ, T_SZ, teW1, teb1, teg, tebt, teW2, teb2, thist);

    // 6) logits + transposes
    logits_kernel<<<4, 256, 0, stream>>>(pf, thf, torg, thist, logit_scale, out);
}
